// InputSpatialTransformer_438086664512
// MI455X (gfx1250) — matrix operand order repaired and checked by reading, not compiled, not run on hardware
//
#include <hip/hip_runtime.h>
#include <hip/hip_bf16.h>

typedef __attribute__((ext_vector_type(16))) _Float16 v16h;
typedef __attribute__((ext_vector_type(8)))  _Float16 v8h;
typedef __attribute__((ext_vector_type(8)))  float    v8f;

#define BN_EPS 1e-5f

// -------- zero the per-layer BN stats region (must happen every call) --------
__global__ void zero_stats_kernel(float* p, int n) {
    int i = blockIdx.x * 256 + threadIdx.x;
    if (i < n) p[i] = 0.f;
}

// ===================== k=11 layers: packed-weight repack ======================
// Wp layout: [COUT][CIN*122][16] f16. Row r = ci*122 + (kd*11+kh); kw padded
// 11->16 with zeros, per-ci rows padded 121->122 with a zero row. The zeros in
// B make the corresponding (stale) A-fragment lanes contribute exactly 0.
template<int CIN, int COUT>
__global__ void repack_w11(const float* __restrict__ w, _Float16* __restrict__ wp)
{
    constexpr int NRP = CIN * 122;
    const int i = blockIdx.x * 256 + threadIdx.x;
    if (i >= COUT * NRP * 16) return;
    const int kw = i & 15;
    const int r  = (i >> 4) % NRP;
    const int co = (i >> 4) / NRP;
    const int ci = r / 122, lr = r - ci * 122;
    float v = 0.f;
    if (lr < 121 && kw < 11) {
        const int kd = lr / 11, kh = lr - kd * 11;
        v = w[(((co * CIN + ci) * 11 + kd) * 11 + kh) * 11 + kw];
    }
    wp[i] = (_Float16)v;
}

// ============== k=11 stride-2 conv via WMMA with LDS row cache ===============
// M-tile = 16 consecutive ow at fixed (od,oh). Per input row (ci,kd,kh) the
// whole tile needs one contiguous 41-elem run -> cached per-wave in LDS.
// A fragment = two contiguous 16B LDS reads; B fragment = one 32B L2-hot load.
// No barriers in the K loop (row cache is wave-private).
template<int CIN, int COUT, int ID, int OD, bool FIRST>
__global__ __launch_bounds__(128)
void conv11_wmma(const void* __restrict__ src, const _Float16* __restrict__ wp,
                 const float* __restrict__ bias, const float* __restrict__ prevStat,
                 _Float16* __restrict__ dst, float* __restrict__ stat)
{
    constexpr int S   = OD * OD * OD;
    constexpr int WT  = (OD + 15) / 16;     // ow tiles
    constexpr int TOT = OD * OD * WT;       // wave-tiles per batch
    constexpr int NRP = CIN * 122;

    __shared__ __align__(16) _Float16 sRow[4][122][48];  // per-wave row cache
    __shared__ float sSum[16];
    __shared__ float sSq[16];

    const int tid  = threadIdx.x;
    const int wave = tid >> 5;
    const int lane = tid & 31;
    const int half = lane >> 4;
    const int mrow = lane & 15;
    const int b    = blockIdx.y;
    const int t    = blockIdx.x * 4 + wave;
    const bool live = (t < TOT);
    const int tt   = live ? t : 0;
    const int od   = tt / (OD * WT);
    const int rem  = tt - od * (OD * WT);
    const int oh   = rem / WT;
    const int ow0  = (rem - oh * WT) * 16;
    const int rlane = lane >> 3;            // 4 rows in flight
    const int jlane = lane & 7;             // 8 lanes sweep a row

    v8f acc = {};

    for (int ci = 0; ci < CIN; ++ci) {
        float sc = 1.f, sh = 0.f;
        if constexpr (!FIRST) { sc = prevStat[32 + ci]; sh = prevStat[48 + ci]; }
        __syncthreads();   // uniform; orders prev K-loop reads vs. restage
        // ---- stage 122 rows x 48 f16 (41 valid taps, rest zero) ----
        for (int row = rlane; row < 122; row += 4) {
            const bool rv = (row < 121);
            int base = 0;
            if (rv) {
                const int kd = row / 11, kh = row - kd * 11;
                base = (((b * CIN + ci) * ID + (od * 2 + kd)) * ID + (oh * 2 + kh)) * ID + ow0 * 2;
            }
            for (int j = jlane; j < 48; j += 8) {
                float f = 0.f;
                if (rv && j < 41 && (ow0 * 2 + j) < ID) {
                    if constexpr (FIRST) {
                        f = ((const float*)src)[base + j];
                    } else {
                        f = (float)((const _Float16*)src)[base + j];
                        f = f * sc + sh;                       // folded BN
                    }
                }
                sRow[wave][row][j] = (_Float16)f;
            }
        }
        // ---- 61 WMMA chunks over this ci (rows 2r2, 2r2+1) ----
        for (int r2 = 0; r2 < 61; ++r2) {
            v8h alo, ahi;
            __builtin_memcpy(&alo, &sRow[wave][2 * r2 + half][2 * mrow    ], 16);
            __builtin_memcpy(&ahi, &sRow[wave][2 * r2 + half][2 * mrow + 8], 16);
            v16h a = __builtin_shufflevector(alo, ahi,
                         0,1,2,3,4,5,6,7,8,9,10,11,12,13,14,15);
            v16h bf = {};
            if (mrow < COUT)
                bf = *(const v16h*)(wp + ((mrow * NRP + ci * 122 + 2 * r2 + half) << 4));
            acc = __builtin_amdgcn_wmma_f32_16x16x32_f16(
                      false, a, false, bf, (short)0, acc, false, false);
        }
    }

    // ---- epilogue: bias + relu (both k=11 layers), f16 store, BN stats ----
    const float bval = (mrow < COUT) ? bias[mrow] : 0.f;
    float lsum = 0.f, lsq = 0.f;
    const int pb = (od * OD + oh) * OD + ow0;
    #pragma unroll
    for (int r = 0; r < 8; ++r) {
        const int m = r + 8 * half;            // D element M = r + 8*half
        if (live && mrow < COUT && (ow0 + m) < OD) {
            float v = fmaxf(acc[r] + bval, 0.f);
            dst[(b * COUT + mrow) * S + pb + m] = (_Float16)v;
            lsum += v; lsq += v * v;
        }
    }
    if (tid < 16) { sSum[tid] = 0.f; sSq[tid] = 0.f; }
    __syncthreads();
    if (mrow < COUT) { atomicAdd(&sSum[mrow], lsum); atomicAdd(&sSq[mrow], lsq); }
    __syncthreads();
    if (tid < COUT) {
        atomicAdd(&stat[tid],      sSum[tid]);
        atomicAdd(&stat[16 + tid], sSq[tid]);
    }
}

// ========= generic WMMA conv (small tail layers, k=5/3, <0.2 GFLOP) ==========
template<int CIN, int COUT, int KS, int ID, int OD, bool RELU>
__global__ __launch_bounds__(128)
void conv_wmma(const _Float16* __restrict__ src, const float* __restrict__ wgt,
               const float* __restrict__ bias, const float* __restrict__ prevStat,
               _Float16* __restrict__ dst, float* __restrict__ stat)
{
    constexpr int S   = OD * OD * OD;
    constexpr int K2  = KS * KS;
    constexpr int K3  = KS * K2;
    constexpr int KT  = CIN * K3;
    constexpr int NCH = (KT + 31) / 32;

    __shared__ __align__(64) _Float16 sA[4][16][32];
    __shared__ __align__(64) _Float16 sB[16][32];
    __shared__ float sSum[16];
    __shared__ float sSq[16];

    const int tid  = threadIdx.x;
    const int wave = tid >> 5;
    const int lane = tid & 31;
    const int half = lane >> 4;
    const int mrow = lane & 15;
    const int b    = blockIdx.y;
    const int tilebase = (blockIdx.x * 4 + wave) * 16;

    v8f acc = {};
    for (int ch = 0; ch < NCH; ++ch) {
        __syncthreads();
        for (int e = tid; e < 512; e += 128) {
            const int n = e >> 5, kk = e & 31, k = ch * 32 + kk;
            float w = 0.f;
            if (n < COUT && k < KT) w = wgt[n * KT + k];
            sB[n][kk] = (_Float16)w;
        }
        for (int e = lane; e < 512; e += 32) {
            const int m = e >> 5, kk = e & 31, k = ch * 32 + kk;
            const int p = tilebase + m;
            float f = 0.f;
            if (k < KT && p < S) {
                const int ci = k / K3;
                int r  = k - ci * K3;
                const int kd = r / K2;  r -= kd * K2;
                const int kh = r / KS;
                const int kw = r - kh * KS;
                const int od = p / (OD * OD);
                int q  = p - od * OD * OD;
                const int oh = q / OD;
                const int ow = q - oh * OD;
                const int sidx = (((b * CIN + ci) * ID + (od * 2 + kd)) * ID + (oh * 2 + kh)) * ID + (ow * 2 + kw);
                f = (float)src[sidx] * prevStat[32 + ci] + prevStat[48 + ci];
            }
            sA[wave][m][kk] = (_Float16)f;
        }
        __syncthreads();
        const _Float16* arow = &sA[wave][mrow][0];
        v8h alo = *(const v8h*)(arow + 16 * half);
        v8h ahi = *(const v8h*)(arow + 16 * half + 8);
        v16h a  = __builtin_shufflevector(alo, ahi, 0,1,2,3,4,5,6,7,8,9,10,11,12,13,14,15);
        v16h bf = *(const v16h*)(&sB[mrow][16 * half]);
        acc = __builtin_amdgcn_wmma_f32_16x16x32_f16(
                  false, a, false, bf, (short)0, acc, false, false);
    }

    const float bval = (mrow < COUT) ? bias[mrow] : 0.f;
    float lsum = 0.f, lsq = 0.f;
    #pragma unroll
    for (int r = 0; r < 8; ++r) {
        const int p = tilebase + r + 8 * half;
        if (mrow < COUT && p < S) {
            float v = acc[r] + bval;
            if (RELU) v = fmaxf(v, 0.f);
            dst[(b * COUT + mrow) * S + p] = (_Float16)v;
            lsum += v; lsq += v * v;
        }
    }
    if (tid < 16) { sSum[tid] = 0.f; sSq[tid] = 0.f; }
    __syncthreads();
    if (mrow < COUT) { atomicAdd(&sSum[mrow], lsum); atomicAdd(&sSq[mrow], lsq); }
    __syncthreads();
    if (tid < COUT) {
        atomicAdd(&stat[tid],      sSum[tid]);
        atomicAdd(&stat[16 + tid], sSq[tid]);
    }
}

// -------- fold BN: stat[32..47]=scale, stat[48..63]=shift --------
__global__ void bn_finalize(float* stat, const float* g, const float* beta,
                            int cout, float invCount)
{
    int c = threadIdx.x;
    if (c < cout) {
        float mu  = stat[c] * invCount;
        float var = stat[16 + c] * invCount - mu * mu;
        float s   = g[c] * rsqrtf(var + BN_EPS);
        stat[32 + c] = s;
        stat[48 + c] = beta[c] - mu * s;
    }
}

// -------- tiny MLP head: flatten(2x112) -> fc1(relu) -> fc2 -> theta (2x12) --------
__global__ __launch_bounds__(128)
void fc_theta(const _Float16* __restrict__ y6, const float* __restrict__ stat6,
              const float* __restrict__ fc1w, const float* __restrict__ fc1b,
              const float* __restrict__ fc2w, const float* __restrict__ fc2b,
              float* __restrict__ theta)
{
    __shared__ float feat[2][112];
    __shared__ float hid[2][32];
    const int t = threadIdx.x;
    for (int e = t; e < 224; e += 128) {
        const int b = e / 112, i = e - b * 112, c = i >> 3, s = i & 7;
        feat[b][i] = (float)y6[(b * 14 + c) * 8 + s] * stat6[32 + c] + stat6[48 + c];
    }
    __syncthreads();
    if (t < 64) {
        const int b = t >> 5, j = t & 31;
        float s = fc1b[j];
        #pragma unroll 4
        for (int i = 0; i < 112; ++i) s += feat[b][i] * fc1w[j * 112 + i];
        hid[b][j] = fmaxf(s, 0.f);
    }
    __syncthreads();
    if (t < 24) {
        const int b = t / 12, i = t - b * 12;
        float s = fc2b[i];
        #pragma unroll
        for (int j = 0; j < 32; ++j) s += hid[b][j] * fc2w[i * 32 + j];
        theta[b * 12 + i] = s;
    }
}

// -------- affine_grid + trilinear grid_sample (zeros, align_corners=False) --------
__global__ __launch_bounds__(256)
void sampler(const float* __restrict__ x, const float* __restrict__ theta,
             float* __restrict__ out)
{
    constexpr int N = 224, NN = N * N, NNN = N * NN;
    const int b   = blockIdx.y;
    const int idx = blockIdx.x * 256 + threadIdx.x;
    const int ow = idx % N;
    const int oh = (idx / N) % N;
    const int od = idx / NN;
    const float inv = 1.f / (float)N;
    const float cx = (2.f * ow + 1.f) * inv - 1.f;
    const float cy = (2.f * oh + 1.f) * inv - 1.f;
    const float cz = (2.f * od + 1.f) * inv - 1.f;
    const float* th = theta + b * 12;
    const float gx = fmaf(th[0], cx, fmaf(th[1],  cy, fmaf(th[2],  cz, th[3])));
    const float gy = fmaf(th[4], cx, fmaf(th[5],  cy, fmaf(th[6],  cz, th[7])));
    const float gz = fmaf(th[8], cx, fmaf(th[9],  cy, fmaf(th[10], cz, th[11])));
    const float ix = ((gx + 1.f) * (float)N - 1.f) * 0.5f;
    const float iy = ((gy + 1.f) * (float)N - 1.f) * 0.5f;
    const float iz = ((gz + 1.f) * (float)N - 1.f) * 0.5f;
    const float x0f = floorf(ix), y0f = floorf(iy), z0f = floorf(iz);
    const float fx = ix - x0f, fy = iy - y0f, fz = iz - z0f;
    const int x0 = (int)x0f, y0 = (int)y0f, z0 = (int)z0f;
    const float* vb = x + (size_t)b * NNN;
    {
        int pz = min(max(z0, 0), N - 1), py = min(max(y0, 0), N - 1), px = min(max(x0, 0), N - 1);
        __builtin_prefetch(vb + (pz * N + py) * N + px, 0, 1);   // global_prefetch_b8
    }
    float acc = 0.f;
    #pragma unroll
    for (int c = 0; c < 8; ++c) {
        const int dx = c & 1, dy = (c >> 1) & 1, dz = c >> 2;
        const int xi = x0 + dx, yi = y0 + dy, zi = z0 + dz;
        const float w = (dx ? fx : 1.f - fx) * (dy ? fy : 1.f - fy) * (dz ? fz : 1.f - fz);
        if (xi >= 0 && xi < N && yi >= 0 && yi < N && zi >= 0 && zi < N)
            acc += w * vb[(zi * N + yi) * N + xi];
    }
    __builtin_nontemporal_store(acc, &out[(size_t)b * NNN + idx]);
}

extern "C" void kernel_launch(void* const* d_in, const int* in_sizes, int n_in,
                              void* d_out, int out_size, void* d_ws, size_t ws_size,
                              hipStream_t stream)
{
    const float* x = (const float*)d_in[0];
    const float *cw[6], *cb[6], *bg[6], *bb[6];
    for (int i = 0; i < 6; ++i) {
        cw[i] = (const float*)d_in[1 + i];
        cb[i] = (const float*)d_in[7 + i];
        bg[i] = (const float*)d_in[13 + i];
        bb[i] = (const float*)d_in[19 + i];
    }
    const float* fc1w = (const float*)d_in[25];
    const float* fc1b = (const float*)d_in[26];
    const float* fc2w = (const float*)d_in[27];
    const float* fc2b = (const float*)d_in[28];

    // ---- workspace carve-up ----
    char*  ws  = (char*)d_ws;
    size_t off = 0;
    auto take = [&](size_t bytes) -> char* {
        char* p = ws + off;
        off = (off + bytes + 255) & ~(size_t)255;
        return p;
    };
    float*    theta = (float*)take(24 * sizeof(float));
    float*    stats = (float*)take(6 * 64 * sizeof(float));      // layer L at stats + L*64
    _Float16* wp0   = (_Float16*)take((size_t)4 * 122 * 16 * 2); // packed k=11 weights, L0
    _Float16* wp1   = (_Float16*)take((size_t)6 * 4 * 122 * 16 * 2); // L1
    static const long long YE[6] = {
        2LL * 4 * 107 * 107 * 107, 2LL * 6 * 49 * 49 * 49, 2LL * 8 * 23 * 23 * 23,
        2LL * 10 * 11 * 11 * 11,   2LL * 12 * 5 * 5 * 5,   2LL * 14 * 2 * 2 * 2 };
    _Float16* y[6];
    for (int i = 0; i < 6; ++i) y[i] = (_Float16*)take((size_t)YE[i] * 2);

    zero_stats_kernel<<<2, 256, 0, stream>>>(stats, 384);
    repack_w11<1, 4><<<(4 * 122 * 16 + 255) / 256, 256, 0, stream>>>(cw[0], wp0);
    repack_w11<4, 6><<<(6 * 4 * 122 * 16 + 255) / 256, 256, 0, stream>>>(cw[1], wp1);

    // ---- conv tower: two WMMA-dense k=11 layers, generic WMMA tail ----
    { constexpr int S = 107*107*107, TOT = 107*107*7;
      conv11_wmma<1,4,224,107,true><<<dim3((TOT+3)/4, 2), 128, 0, stream>>>(
          x, wp0, cb[0], nullptr, y[0], stats + 0*64);
      bn_finalize<<<1, 16, 0, stream>>>(stats + 0*64, bg[0], bb[0], 4, 1.f/(2.f*S)); }
    { constexpr int S = 49*49*49, TOT = 49*49*4;
      conv11_wmma<4,6,107,49,false><<<dim3((TOT+3)/4, 2), 128, 0, stream>>>(
          y[0], wp1, cb[1], stats + 0*64, y[1], stats + 1*64);
      bn_finalize<<<1, 16, 0, stream>>>(stats + 1*64, bg[1], bb[1], 6, 1.f/(2.f*S)); }
    { constexpr int S = 23*23*23;
      conv_wmma<6,8,5,49,23,true><<<dim3((S+63)/64, 2), 128, 0, stream>>>(
          y[1], cw[2], cb[2], stats + 1*64, y[2], stats + 2*64);
      bn_finalize<<<1, 16, 0, stream>>>(stats + 2*64, bg[2], bb[2], 8, 1.f/(2.f*S)); }
    { constexpr int S = 11*11*11;
      conv_wmma<8,10,3,23,11,true><<<dim3((S+63)/64, 2), 128, 0, stream>>>(
          y[2], cw[3], cb[3], stats + 2*64, y[3], stats + 3*64);
      bn_finalize<<<1, 16, 0, stream>>>(stats + 3*64, bg[3], bb[3], 10, 1.f/(2.f*S)); }
    { constexpr int S = 5*5*5;
      conv_wmma<10,12,3,11,5,true><<<dim3((S+63)/64, 2), 128, 0, stream>>>(
          y[3], cw[4], cb[4], stats + 3*64, y[4], stats + 4*64);
      bn_finalize<<<1, 16, 0, stream>>>(stats + 4*64, bg[4], bb[4], 12, 1.f/(2.f*S)); }
    { constexpr int S = 2*2*2;
      conv_wmma<12,14,3,5,2,false><<<dim3((S+63)/64, 2), 128, 0, stream>>>(
          y[4], cw[5], cb[5], stats + 4*64, y[5], stats + 5*64);
      bn_finalize<<<1, 16, 0, stream>>>(stats + 5*64, bg[5], bb[5], 14, 1.f/(2.f*S)); }

    // ---- MLP head -> theta ----
    fc_theta<<<1, 128, 0, stream>>>(y[5], stats + 5*64, fc1w, fc1b, fc2w, fc2b, theta);

    // ---- affine grid + trilinear sample (HBM-bound, ~180 MB @ 23.3 TB/s) ----
    constexpr int NNN = 224 * 224 * 224;
    sampler<<<dim3(NNN / 256, 2), 256, 0, stream>>>(x, theta, (float*)d_out);
}
